// LSTMSlider_59261958750775
// MI455X (gfx1250) — compile-verified, matrix operand order checked by reading, not run on hardware
//
#include <hip/hip_runtime.h>
#include <cstdint>

// ---------------------------------------------------------------------------
// Problem constants (match reference)
// ---------------------------------------------------------------------------
#define SEQ    4096
#define C_IN   256
#define KWIN   128     // slice length (batch dim of LSTM)
#define STRIDE 32
#define OUTD   256
#define DKWIN  25
#define NSLICE 125     // (SEQ-KWIN)/STRIDE + 1
#define NSTEP  124     // scan consumes slices[1:]

typedef __bf16 bf16_t;
typedef __attribute__((ext_vector_type(16))) __bf16 v16bf;
typedef __attribute__((ext_vector_type(8)))  float  v8f;

union frag_u { unsigned int u[8]; v16bf v; };

__device__ __forceinline__ unsigned short f2bf(float x) {
    unsigned int u = __builtin_bit_cast(unsigned int, x);
    u += 0x7FFFu + ((u >> 16) & 1u);          // round-to-nearest-even
    return (unsigned short)(u >> 16);
}

__device__ __forceinline__ float sigmoidf_(float x) {
    return 1.0f / (1.0f + __expf(-x));
}

// K-offset inside a 32-wide K window for VGPR v of the 16-bit A/B WMMA layout
// (ISA 7.12.2: lanes 0-15 hold K pairs {0..7,16..23}, lanes 16-31 {8..15,24..31})
__device__ __forceinline__ int wmma_k_of(int v, int hi) {
    return ((v < 4) ? (2 * v) : (16 + 2 * (v - 4))) + 8 * hi;
}

// Load a 16x32 bf16 A fragment from a row-major [rows x 256] bf16 matrix.
__device__ __forceinline__ v16bf load_A_rowmajor(const unsigned short* base,
                                                 int row0, int k0) {
    const int lane = threadIdx.x & 31;
    const int m    = lane & 15;
    const int hi   = lane >> 4;
    const unsigned char* p =
        (const unsigned char*)(base + (size_t)(row0 + m) * 256);
    frag_u f;
#pragma unroll
    for (int v = 0; v < 8; ++v) {
        int k = k0 + wmma_k_of(v, hi);
        f.u[v] = *(const unsigned int*)(p + (size_t)k * 2);  // pair (k,k+1), 4B aligned
    }
    return f.v;
}

// Load a pre-swizzled 32x16 B fragment: tile stored as [32 lanes][8 dwords].
__device__ __forceinline__ v16bf load_B_packed(const unsigned int* tile) {
    const int lane = threadIdx.x & 31;
    const uint4* q = (const uint4*)(tile + lane * 8);
    uint4 a = q[0], b = q[1];
    frag_u f;
    f.u[0] = a.x; f.u[1] = a.y; f.u[2] = a.z; f.u[3] = a.w;
    f.u[4] = b.x; f.u[5] = b.y; f.u[6] = b.z; f.u[7] = b.w;
    return f.v;
}

__device__ __forceinline__ v8f wmma_bf16(v16bf a, v16bf b, v8f c) {
    return __builtin_amdgcn_wmma_f32_16x16x32_bf16(
        false, a, false, b, (short)0, c, false, false);
}

// ---------------------------------------------------------------------------
// K1: moving average (replicate pad, window 25) -> trend/seas in bf16
// ---------------------------------------------------------------------------
__global__ __launch_bounds__(256) void k_prep(const float* __restrict__ path,
                                              unsigned short* __restrict__ trend_bf,
                                              unsigned short* __restrict__ seas_bf) {
    const int t = blockIdx.x;      // 0..4095
    const int c = threadIdx.x;     // 0..255
    float s = 0.f;
#pragma unroll
    for (int o = -12; o <= 12; ++o) {
        int tt = t + o;
        tt = tt < 0 ? 0 : (tt > SEQ - 1 ? SEQ - 1 : tt);
        s += path[(size_t)tt * C_IN + c];
    }
    const float tr = s * (1.0f / (float)DKWIN);
    const float se = path[(size_t)t * C_IN + c] - tr;
    trend_bf[(size_t)t * C_IN + c] = f2bf(tr);
    seas_bf [(size_t)t * C_IN + c] = f2bf(se);
}

// ---------------------------------------------------------------------------
// K2a: pack projection weights into WMMA-B tile layout.
//   packedPB[which][kt(8)][nt(16)][lane(32)][v(8)] : B[k=c][n=e] = W[e][c]
// ---------------------------------------------------------------------------
__global__ __launch_bounds__(256) void k_pack_projB(const float* __restrict__ trend_W,
                                                    const float* __restrict__ seas_W,
                                                    unsigned int* __restrict__ packedPB) {
    const unsigned i = blockIdx.x * 256u + threadIdx.x;  // 0..65535
    const int v     = i & 7;
    const int lane  = (i >> 3) & 31;
    const int nt    = (i >> 8) & 15;
    const int kt    = (i >> 12) & 7;
    const int which = (i >> 15) & 1;
    const int hi = lane >> 4;
    const int e  = nt * 16 + (lane & 15);
    const int k  = kt * 32 + wmma_k_of(v, hi);
    const float* W = which ? seas_W : trend_W;           // [e][c]
    const float w0 = W[(size_t)e * C_IN + k];
    const float w1 = W[(size_t)e * C_IN + k + 1];
    packedPB[i] = (unsigned int)f2bf(w0) | ((unsigned int)f2bf(w1) << 16);
}

// ---------------------------------------------------------------------------
// K2b: pack LSTM weights. Combined K=512 (k<256 -> hW, k>=256 -> uW).
//   packedGB[dir(2)][g(4)][kt(16)][nt(16)][lane(32)][v(8)]
//   B[k][n=e] = (k<256 ? hW[g][e][k] : uW[g][e][k-256])
// ---------------------------------------------------------------------------
__global__ __launch_bounds__(256) void k_pack_gateB(const float* __restrict__ fwd_hW,
                                                    const float* __restrict__ fwd_uW,
                                                    const float* __restrict__ back_hW,
                                                    const float* __restrict__ back_uW,
                                                    unsigned int* __restrict__ packedGB) {
    const unsigned i = blockIdx.x * 256u + threadIdx.x;  // 0..524287
    const int v    = i & 7;
    const int lane = (i >> 3) & 31;
    const int nt   = (i >> 8) & 15;
    const int kt   = (i >> 12) & 15;
    const int g    = (i >> 16) & 3;
    const int dir  = (i >> 18) & 1;
    const int hi = lane >> 4;
    const int e  = nt * 16 + (lane & 15);
    const int k  = kt * 32 + wmma_k_of(v, hi);
    const float* hW = dir ? back_hW : fwd_hW;            // [4][e][d]
    const float* uW = dir ? back_uW : fwd_uW;
    float w0, w1;
    if (k < 256) {
        w0 = hW[((size_t)g * OUTD + e) * OUTD + k];
        w1 = hW[((size_t)g * OUTD + e) * OUTD + k + 1];
    } else {
        w0 = uW[((size_t)g * OUTD + e) * OUTD + (k - 256)];
        w1 = uW[((size_t)g * OUTD + e) * OUTD + (k - 255)];
    }
    packedGB[i] = (unsigned int)f2bf(w0) | ((unsigned int)f2bf(w1) << 16);
}

// ---------------------------------------------------------------------------
// K2c: combined biases. pb[e] = trend_b+seas_b ; gb[dir][g][e] = w_b + u_b
// ---------------------------------------------------------------------------
__global__ __launch_bounds__(256) void k_pack_bias(const float* __restrict__ trend_b,
                                                   const float* __restrict__ seas_b,
                                                   const float* __restrict__ fwd_w_b,
                                                   const float* __restrict__ fwd_u_b,
                                                   const float* __restrict__ back_w_b,
                                                   const float* __restrict__ back_u_b,
                                                   float* __restrict__ pb,
                                                   float* __restrict__ gb) {
    const int e = threadIdx.x;  // 0..255
    pb[e] = trend_b[e] + seas_b[e];
#pragma unroll
    for (int g = 0; g < 4; ++g) {
        gb[(0 * 4 + g) * OUTD + e] = fwd_w_b[g * OUTD + e] + fwd_u_b[g * OUTD + e];
        gb[(1 * 4 + g) * OUTD + e] = back_w_b[g * OUTD + e] + back_u_b[g * OUTD + e];
    }
}

// ---------------------------------------------------------------------------
// K3: projection GEMM. proj[t,e] = trend@trend_W^T + seas@seas_W^T + pb
//     4096 waves, 1 wave = one 16x16 output tile, K=256 per source.
// ---------------------------------------------------------------------------
__global__ __launch_bounds__(256) void k_proj(const unsigned short* __restrict__ trend_bf,
                                              const unsigned short* __restrict__ seas_bf,
                                              const unsigned int* __restrict__ packedPB,
                                              const float* __restrict__ pb,
                                              unsigned short* __restrict__ proj_bf) {
    const int wave  = (int)((blockIdx.x * blockDim.x + threadIdx.x) >> 5); // 0..4095
    const int mtile = wave >> 4;    // 0..255  (rows of proj, 16 at a time)
    const int ntile = wave & 15;    // 0..15   (cols, 16 at a time)
    const int lane  = threadIdx.x & 31;

    v8f acc = {};
#pragma unroll
    for (int src = 0; src < 2; ++src) {
        const unsigned short* A = src ? seas_bf : trend_bf;
#pragma unroll
        for (int kt = 0; kt < 8; ++kt) {
            v16bf a = load_A_rowmajor(A, mtile * 16, kt * 32);
            const unsigned int* tb =
                packedPB + ((((size_t)src * 8 + kt) * 16 + ntile) * 32) * 8;
            v16bf b = load_B_packed(tb);
            acc = wmma_bf16(a, b, acc);
        }
    }
    const int e    = ntile * 16 + (lane & 15);
    const int hi   = lane >> 4;
    const float bias = pb[e];
#pragma unroll
    for (int r = 0; r < 8; ++r) {
        const int m = mtile * 16 + r + 8 * hi;
        proj_bf[(size_t)m * OUTD + e] = f2bf(acc[r] + bias);
    }
}

// ---------------------------------------------------------------------------
// K4: persistent bi-directional LSTM. blockIdx.x = direction (0=fwd, 1=back).
//     1024 threads = 32 waves; wave owns 4 fixed 16x16 tiles of the [128,256]
//     hidden state -> cell state lives in registers across all 124 steps.
//     h double-buffered in LDS as bf16 (2 * 128 * 256 * 2B = 128 KB).
//     Per step per tile: G = [h ; x] @ [hW ; uW]  (K=512, 4 gates).
// ---------------------------------------------------------------------------
extern __shared__ unsigned short s_h[];   // [2][128][256] bf16

__global__ __launch_bounds__(1024, 1) void k_lstm(const unsigned short* __restrict__ proj_bf,
                                                  const unsigned int* __restrict__ packedGB,
                                                  const float* __restrict__ gb,
                                                  float* __restrict__ out) {
    const int dir  = blockIdx.x;
    const int tid  = threadIdx.x;
    const int lane = tid & 31;
    const int wv   = tid >> 5;          // 0..31
    const int hi   = lane >> 4;

    // zero both h buffers (h0 = 0)
    for (int i = tid; i < 2 * KWIN * OUTD; i += 1024) s_h[i] = 0;

    // per-owned-tile combined biases (depend only on column e)
    float bias[4][4];
#pragma unroll
    for (int j = 0; j < 4; ++j) {
        const int tileId = wv * 4 + j;
        const int e = (tileId & 15) * 16 + (lane & 15);
#pragma unroll
        for (int g = 0; g < 4; ++g)
            bias[j][g] = gb[((size_t)dir * 4 + g) * OUTD + e];
    }

    float c_state[4][8];
#pragma unroll
    for (int j = 0; j < 4; ++j)
#pragma unroll
        for (int r = 0; r < 8; ++r) c_state[j][r] = 0.f;

    __syncthreads();

    int cur = 0;
    for (int t = 0; t < NSTEP; ++t) {
        const int sidx = dir == 0 ? (1 + t) : (123 - t);
        const unsigned short* xbase = proj_bf + (size_t)sidx * STRIDE * OUTD;

        // prefetch next step's x slice (64 KB spread over 1024 lanes)
        if (t + 1 < NSTEP) {
            const int ns = dir == 0 ? (sidx + 1) : (sidx - 1);
            const unsigned char* pf =
                (const unsigned char*)(proj_bf + (size_t)ns * STRIDE * OUTD);
            __builtin_prefetch(pf + (size_t)tid * 64, 0, 0);
        }

        const unsigned short* hsrc = s_h + (size_t)cur * KWIN * OUTD;
        unsigned short*       hdst = s_h + (size_t)(cur ^ 1) * KWIN * OUTD;

#pragma unroll
        for (int j = 0; j < 4; ++j) {
            const int tileId = wv * 4 + j;
            const int mtile  = tileId >> 4;   // 0..7
            const int ntile  = tileId & 15;   // 0..15

            v8f acc[4] = {};
            for (int kt = 0; kt < 16; ++kt) {
                v16bf a = (kt < 8)
                    ? load_A_rowmajor(hsrc, mtile * 16, kt * 32)          // h part
                    : load_A_rowmajor(xbase, mtile * 16, (kt - 8) * 32);  // x part
#pragma unroll
                for (int g = 0; g < 4; ++g) {
                    const unsigned int* tb = packedGB +
                        (((((size_t)dir * 4 + g) * 16 + kt) * 16 + ntile) * 32) * 8;
                    v16bf b = load_B_packed(tb);
                    acc[g] = wmma_bf16(a, b, acc[g]);
                }
            }

            // gate order: f, i, o, c_tilde (matches reference)
            const int e = ntile * 16 + (lane & 15);
#pragma unroll
            for (int r = 0; r < 8; ++r) {
                const float fg = sigmoidf_(acc[0][r] + bias[j][0]);
                const float ig = sigmoidf_(acc[1][r] + bias[j][1]);
                const float og = sigmoidf_(acc[2][r] + bias[j][2]);
                const float ct = tanhf(acc[3][r] + bias[j][3]);
                // non-standard: sigmoid on the cell update (per reference)
                const float cn = sigmoidf_(fg * c_state[j][r] + ig * ct);
                const float hn = og * tanhf(cn);
                c_state[j][r] = cn;
                const int m = mtile * 16 + r + 8 * hi;
                hdst[(size_t)m * OUTD + e] = f2bf(hn);
                if (t == NSTEP - 1)
                    out[(size_t)m * (2 * OUTD) + dir * OUTD + e] = hn;
            }
        }
        __syncthreads();
        cur ^= 1;
    }
}

// ---------------------------------------------------------------------------
// Host launcher
// ---------------------------------------------------------------------------
extern "C" void kernel_launch(void* const* d_in, const int* in_sizes, int n_in,
                              void* d_out, int out_size, void* d_ws, size_t ws_size,
                              hipStream_t stream) {
    (void)in_sizes; (void)n_in; (void)out_size; (void)ws_size;

    const float* path     = (const float*)d_in[0];
    const float* trend_W  = (const float*)d_in[1];
    const float* trend_b  = (const float*)d_in[2];
    const float* seas_W   = (const float*)d_in[3];
    const float* seas_b   = (const float*)d_in[4];
    const float* fwd_w_W  = (const float*)d_in[5];
    const float* fwd_w_b  = (const float*)d_in[6];
    const float* fwd_u_W  = (const float*)d_in[7];
    const float* fwd_u_b  = (const float*)d_in[8];
    const float* back_w_W = (const float*)d_in[9];
    const float* back_w_b = (const float*)d_in[10];
    const float* back_u_W = (const float*)d_in[11];
    const float* back_u_b = (const float*)d_in[12];
    float* out = (float*)d_out;

    // workspace layout
    char* ws = (char*)d_ws;
    unsigned short* trend_bf = (unsigned short*)(ws + 0);          // 2 MB
    unsigned short* seas_bf  = (unsigned short*)(ws + 2097152);    // 2 MB
    unsigned short* proj_bf  = (unsigned short*)(ws + 4194304);    // 2 MB
    unsigned int*   packedPB = (unsigned int*)  (ws + 6291456);    // 256 KB
    unsigned int*   packedGB = (unsigned int*)  (ws + 6553600);    // 2 MB
    float*          pb       = (float*)         (ws + 8650752);    // 1 KB
    float*          gb       = (float*)         (ws + 8651776);    // 8 KB

    hipLaunchKernelGGL(k_prep, dim3(SEQ), dim3(C_IN), 0, stream,
                       path, trend_bf, seas_bf);
    hipLaunchKernelGGL(k_pack_projB, dim3(256), dim3(256), 0, stream,
                       trend_W, seas_W, packedPB);
    hipLaunchKernelGGL(k_pack_gateB, dim3(2048), dim3(256), 0, stream,
                       fwd_w_W, fwd_u_W, back_w_W, back_u_W, packedGB);
    hipLaunchKernelGGL(k_pack_bias, dim3(1), dim3(256), 0, stream,
                       trend_b, seas_b, fwd_w_b, fwd_u_b, back_w_b, back_u_b,
                       pb, gb);
    hipLaunchKernelGGL(k_proj, dim3(512), dim3(256), 0, stream,
                       trend_bf, seas_bf, packedPB, pb, proj_bf);

    // allow >64KB dynamic LDS (gfx1250 WGP has 320 KB); ignore failures
    (void)hipFuncSetAttribute((const void*)k_lstm,
                              hipFuncAttributeMaxDynamicSharedMemorySize,
                              2 * KWIN * OUTD * (int)sizeof(unsigned short));

    hipLaunchKernelGGL(k_lstm, dim3(2), dim3(1024),
                       2 * KWIN * OUTD * sizeof(unsigned short), stream,
                       proj_bf, packedGB, gb, out);
}
